// TemporalAttention_68564857913909
// MI455X (gfx1250) — matrix operand order repaired and checked by reading, not compiled, not run on hardware
//
#include <hip/hip_runtime.h>

typedef unsigned short u16;
typedef __attribute__((ext_vector_type(16))) __bf16 v16bf;
typedef __attribute__((ext_vector_type(8)))  float  v8f;
typedef __attribute__((ext_vector_type(4)))  unsigned int v4u;
typedef __attribute__((ext_vector_type(8)))  int v8i;
typedef __attribute__((ext_vector_type(4)))  int v4i;

union FragB16 { v16bf v; uint4 q[2]; u16 s[16]; };
union FragF32 { v8f v; float f[8]; };

__device__ __forceinline__ u16 f2bf(float x) {
  union { float f; unsigned u; } c; c.f = x;
  unsigned r = c.u + 0x7FFFu + ((c.u >> 16) & 1u);
  return (u16)(r >> 16);
}

__device__ __forceinline__ unsigned lds_off(const void* p) {
  // flat shared pointer: low 32 bits carry the LDS byte offset
  return (unsigned)(size_t)p;
}

// ---------------------------------------------------------------------------
// TDM 2D tile load: global (row-major, bf16) -> LDS with inline row padding.
// D# per CDNA5 ISA ch.8: group0 = {count, lds_addr, global_addr, type=2},
// group1 = {data_size=2B, pad ctl, tensor dims, tile dims, dim0 stride}.
// pad_interval_code: pad after 2<<code dwords; pad_amount_code: (code+1) dwords.
// ---------------------------------------------------------------------------
__device__ __forceinline__ void tdm_load_2d(unsigned lds_addr, const u16* gptr,
                                            unsigned tensor_d0, unsigned tensor_d1,
                                            unsigned tile_d0, unsigned tile_d1,
                                            unsigned stride0,
                                            unsigned pad_interval_code,
                                            unsigned pad_amount_code) {
  unsigned long long ga = (unsigned long long)(size_t)gptr;
  v4u g0;
  g0[0] = 1u;                                    // count=1, user descriptor
  g0[1] = lds_addr;                              // LDS byte address
  g0[2] = (unsigned)ga;                          // global_addr[31:0]
  g0[3] = (unsigned)(ga >> 32) | (2u << 30);     // global_addr[56:32] | type=2
  v8i g1;
  unsigned flags = (1u << 16)                    // data_size = 2 bytes
                 | (1u << 20)                    // pad_enable
                 | (pad_interval_code << 22)
                 | (pad_amount_code << 25);
  g1[0] = (int)flags;
  g1[1] = (int)((tensor_d0 & 0xffffu) << 16);                            // dim0[15:0] @ bits63:48
  g1[2] = (int)(((tensor_d0 >> 16) & 0xffffu) | ((tensor_d1 & 0xffffu) << 16));
  g1[3] = (int)(((tensor_d1 >> 16) & 0xffffu) | (tile_d0 << 16));
  g1[4] = (int)(tile_d1 & 0xffffu);              // tile_dim1, tile_dim2=0
  g1[5] = (int)stride0;                          // dim0 stride [31:0]
  g1[6] = 0;
  g1[7] = 0;
  v4i z4 = {0, 0, 0, 0};                         // 2D: groups 2/3 unused
  v8i z8 = {0, 0, 0, 0, 0, 0, 0, 0};
  __builtin_amdgcn_tensor_load_to_lds(g0, g1, z4, z4, z8, 0);
}

// ---------------------------------------------------------------------------
// 1) transpose + convert weights: in[K][N] f32 -> out[N][K] bf16 (32x32 tiles)
// ---------------------------------------------------------------------------
__global__ __launch_bounds__(256)
void k_convT(const float* __restrict__ in, u16* __restrict__ out, int K, int N) {
  __shared__ float t[32][33];
  const int kb = blockIdx.x * 32, nb = blockIdx.y * 32;
  const int tx = threadIdx.x & 31, ty = threadIdx.x >> 5;  // 32 x 8
#pragma unroll
  for (int r = 0; r < 32; r += 8)
    t[ty + r][tx] = in[(size_t)(kb + ty + r) * N + nb + tx];
  __syncthreads();
#pragma unroll
  for (int r = 0; r < 32; r += 8)
    out[(size_t)(nb + ty + r) * K + kb + tx] = f2bf(t[tx][ty + r]);
}

// ---------------------------------------------------------------------------
// 2) concat + 'b n h w c -> b c (n h w)' + 0.001*posenc -> fp32 + bf16
// ---------------------------------------------------------------------------
__global__ __launch_bounds__(256)
void k_prep(const float* __restrict__ ref, const float* __restrict__ hd,
            float* __restrict__ xf, u16* __restrict__ xbf) {
  int idx = blockIdx.x * blockDim.x + threadIdx.x;  // 0 .. 4*1024*8192-1
  int b   = idx >> 23;
  int rem = idx & 8388607;
  int cc  = rem >> 13;   // token
  int dd  = rem & 8191;  // feature = ((n*32+h)*32+w)
  int n  = dd >> 10;
  int hh = (dd >> 5) & 31;
  int w  = dd & 31;
  float val;
  if (w < 16) val = ref[(((b*8 + n)*32 + hh)*16 + w) * 1024 + cc];
  else        val = hd [(((b*8 + n)*32 + hh)*16 + (w-16)) * 1024 + cc];
  float freq = __expf(-(2.0f * (float)dd / 8192.0f) * 9.210340371976184f);
  float arg  = (float)cc * freq;
  float pe   = (dd & 1) ? __cosf(arg) : __sinf(arg);
  val += 0.001f * pe;
  xf[idx]  = val;
  xbf[idx] = f2bf(val);
}

// ---------------------------------------------------------------------------
// 3/5) bf16 WMMA GEMM, TDM-staged + double buffered.
//      C[M,N] = A[M,K] @ Bt[N,K]^T. Block 128x128x32, 8 waves, wave 64x32.
// ---------------------------------------------------------------------------
#define GLDK 40   // 32 + 4-dword TDM pad => 80 B rows, 16B-aligned

template <bool BF16OUT>
__global__ __launch_bounds__(256)
void k_wmma_gemm(const u16* __restrict__ A, const u16* __restrict__ Bt,
                 u16* __restrict__ Cb, float* __restrict__ Cf,
                 const float* __restrict__ resid, int M, int N, int K) {
  __shared__ u16 As[2][128 * GLDK];
  __shared__ u16 Bs[2][128 * GLDK];
  const int tid = threadIdx.x, lane = tid & 31, wave = tid >> 5;
  const int waveM = wave >> 2, waveN = wave & 3;
  const int mBase = blockIdx.y * 128, nBase = blockIdx.x * 128;

  FragF32 acc[4][2];
#pragma unroll
  for (int i = 0; i < 4; ++i)
#pragma unroll
    for (int j = 0; j < 2; ++j)
#pragma unroll
      for (int e = 0; e < 8; ++e) acc[i][j].f[e] = 0.0f;

  if (wave == 0) {  // prologue: stage first K-slab via Tensor Data Mover
    tdm_load_2d(lds_off(&As[0][0]), A  + (size_t)mBase * K, K, M, 32, 128, K, 3, 3);
    tdm_load_2d(lds_off(&Bs[0][0]), Bt + (size_t)nBase * K, K, N, 32, 128, K, 3, 3);
  }

  int cur = 0;
  for (int k0 = 0; k0 < K; k0 += 32) {
    if (wave == 0) {
      if (k0 + 32 < K) {   // stage next slab, then retire current pair (in-order)
        tdm_load_2d(lds_off(&As[cur ^ 1][0]), A  + (size_t)mBase * K + k0 + 32,
                    K, M, 32, 128, K, 3, 3);
        tdm_load_2d(lds_off(&Bs[cur ^ 1][0]), Bt + (size_t)nBase * K + k0 + 32,
                    K, N, 32, 128, K, 3, 3);
        __builtin_amdgcn_s_wait_tensorcnt(2);
      } else {
        __builtin_amdgcn_s_wait_tensorcnt(0);
      }
    }
    __syncthreads();

    const u16* Ac = &As[cur][0];
    const u16* Bc = &Bs[cur][0];
    FragB16 af[4], bf[2];
#pragma unroll
    for (int mt = 0; mt < 4; ++mt) {           // A frag: K halves split by lane/16
      int row = waveM * 64 + mt * 16 + (lane & 15);
      int kh  = (lane >> 4) * 8;
      af[mt].q[0] = *(const uint4*)&Ac[row * GLDK + kh];
      af[mt].q[1] = *(const uint4*)&Ac[row * GLDK + kh + 16];
    }
#pragma unroll
    for (int nt = 0; nt < 2; ++nt) {           // B frag: K halves split by lane/16
      int n  = waveN * 32 + nt * 16 + (lane & 15);
      int kb = (lane >> 4) * 8;
      bf[nt].q[0] = *(const uint4*)&Bc[n * GLDK + kb];
      bf[nt].q[1] = *(const uint4*)&Bc[n * GLDK + kb + 16];
    }
#pragma unroll
    for (int mt = 0; mt < 4; ++mt)
#pragma unroll
      for (int nt = 0; nt < 2; ++nt)
        acc[mt][nt].v = __builtin_amdgcn_wmma_f32_16x16x32_bf16(
            false, af[mt].v, false, bf[nt].v, (short)0, acc[mt][nt].v, false, false);
    __syncthreads();
    cur ^= 1;
  }

#pragma unroll
  for (int mt = 0; mt < 4; ++mt)
#pragma unroll
    for (int nt = 0; nt < 2; ++nt) {
      int col = nBase + waveN * 32 + nt * 16 + (lane & 15);
      int rb  = mBase + waveM * 64 + mt * 16 + ((lane >> 4) << 3);
#pragma unroll
      for (int e = 0; e < 8; ++e) {
        size_t idx = (size_t)(rb + e) * N + col;
        if (BF16OUT) Cb[idx] = f2bf(acc[mt][nt].f[e]);
        else         Cf[idx] = acc[mt][nt].f[e] + resid[idx];
      }
    }
}

// ---------------------------------------------------------------------------
// 4) Fused flash attention; Q/K tiles staged by TDM, V transposed manually.
//    One block = (batch, head, 32-query tile); 4 waves, wave owns 64 d_v cols.
// ---------------------------------------------------------------------------
#define ALD 264   // 256 + 4-dword TDM pad => 528 B rows, 16B-aligned
#define PLD 40

__global__ __launch_bounds__(128)
void k_flash_attn(const u16* __restrict__ Qg, const u16* __restrict__ Kg,
                  const u16* __restrict__ Vg, u16* __restrict__ Zg) {
  __shared__ u16  Qs[32 * ALD];
  __shared__ u16  Ks[32 * ALD];
  __shared__ u16  Vt[256 * PLD];     // transposed: Vt[d][key]
  __shared__ float Sf[32 * 33];
  __shared__ u16  Ps[32 * PLD];
  __shared__ float rowM[32], rowL[32], rowC[32];

  const int tid = threadIdx.x, lane = tid & 31, wave = tid >> 5;
  const int qbase = blockIdx.x * 32;
  const int head  = blockIdx.y;
  const int batch = blockIdx.z;
  const size_t ld = 2048;
  const u16* Qp = Qg + (size_t)batch * 1024 * ld + head * 256;
  const u16* Kp = Kg + (size_t)batch * 1024 * ld + head * 256;
  const u16* Vp = Vg + (size_t)batch * 1024 * ld + head * 256;

  if (wave == 0)  // stage Q tile [32 x 256] (row stride 2048) via TDM
    tdm_load_2d(lds_off(&Qs[0]), Qp + (size_t)qbase * ld, 2048, 1024, 256, 32,
                2048, 6, 3);
  if (tid < 32) { rowM[tid] = -1e30f; rowL[tid] = 0.0f; }

  FragF32 o[2][4];
#pragma unroll
  for (int mt = 0; mt < 2; ++mt)
#pragma unroll
    for (int nt = 0; nt < 4; ++nt)
#pragma unroll
      for (int e = 0; e < 8; ++e) o[mt][nt].f[e] = 0.0f;

  for (int kb = 0; kb < 32; ++kb) {
    if (wave == 0)  // stage K tile via TDM
      tdm_load_2d(lds_off(&Ks[0]), Kp + (size_t)(kb * 32) * ld, 2048, 1024, 256,
                  32, 2048, 6, 3);
    // V tile transposed into Vt[d][key] (TDM cannot transpose)
#pragma unroll
    for (int r = 0; r < 8; ++r) {
      int row = r * 4 + wave;        // key within block
      int col = lane * 8;            // d
      uint4 vv = *(const uint4*)(Vp + (size_t)(kb * 32 + row) * ld + col);
      const u16* vp = (const u16*)&vv;
#pragma unroll
      for (int i = 0; i < 8; ++i) Vt[(col + i) * PLD + row] = vp[i];
    }
    if (wave == 0) __builtin_amdgcn_s_wait_tensorcnt(0);
    __syncthreads();

    // S = (Q/16) K^T : each wave one 16x16 tile of the 32x32 block
    {
      FragF32 s;
#pragma unroll
      for (int e = 0; e < 8; ++e) s.f[e] = 0.0f;
      int smt = (wave >> 1) * 16, snt = (wave & 1) * 16;
#pragma unroll
      for (int kc = 0; kc < 8; ++kc) {
        FragB16 a, b;
        int qr = smt + (lane & 15);
        int kh = kc * 32 + (lane >> 4) * 8;
        a.q[0] = *(const uint4*)&Qs[qr * ALD + kh];
        a.q[1] = *(const uint4*)&Qs[qr * ALD + kh + 16];
        int kr = snt + (lane & 15);
        int kk = kc * 32 + (lane >> 4) * 8;
        b.q[0] = *(const uint4*)&Ks[kr * ALD + kk];
        b.q[1] = *(const uint4*)&Ks[kr * ALD + kk + 16];
        s.v = __builtin_amdgcn_wmma_f32_16x16x32_bf16(
            false, a.v, false, b.v, (short)0, s.v, false, false);
      }
#pragma unroll
      for (int e = 0; e < 8; ++e)
        Sf[(smt + e + ((lane >> 4) << 3)) * 33 + snt + (lane & 15)] = s.f[e] * 0.0625f;
    }
    __syncthreads();

    // running softmax, one thread per query row
    if (tid < 32) {
      float mOld = rowM[tid], mx = mOld;
#pragma unroll
      for (int j = 0; j < 32; ++j) mx = fmaxf(mx, Sf[tid * 33 + j]);
      float corr = __expf(mOld - mx);
      float l = rowL[tid] * corr;
#pragma unroll
      for (int j = 0; j < 32; ++j) {
        float p = __expf(Sf[tid * 33 + j] - mx);
        l += p;
        Ps[tid * PLD + j] = f2bf(p);
      }
      rowM[tid] = mx; rowL[tid] = l; rowC[tid] = corr;
    }
    __syncthreads();

    // O = O*corr + P @ V
    FragB16 pa[2];
#pragma unroll
    for (int mt = 0; mt < 2; ++mt) {
      int row = mt * 16 + (lane & 15);
      int kh  = (lane >> 4) * 8;
      pa[mt].q[0] = *(const uint4*)&Ps[row * PLD + kh];
      pa[mt].q[1] = *(const uint4*)&Ps[row * PLD + kh + 16];
    }
#pragma unroll
    for (int nt = 0; nt < 4; ++nt) {
      FragB16 vb;
      int d  = wave * 64 + nt * 16 + (lane & 15);
      int kk = (lane >> 4) * 8;
      vb.q[0] = *(const uint4*)&Vt[d * PLD + kk];
      vb.q[1] = *(const uint4*)&Vt[d * PLD + kk + 16];
#pragma unroll
      for (int mt = 0; mt < 2; ++mt) {
#pragma unroll
        for (int e = 0; e < 8; ++e)
          o[mt][nt].f[e] *= rowC[mt * 16 + e + ((lane >> 4) << 3)];
        o[mt][nt].v = __builtin_amdgcn_wmma_f32_16x16x32_bf16(
            false, pa[mt].v, false, vb.v, (short)0, o[mt][nt].v, false, false);
      }
    }
    __syncthreads();
  }

  // normalize and store Z bf16 [4096, 2048]
#pragma unroll
  for (int mt = 0; mt < 2; ++mt)
#pragma unroll
    for (int nt = 0; nt < 4; ++nt) {
      int d = wave * 64 + nt * 16 + (lane & 15);
#pragma unroll
      for (int e = 0; e < 8; ++e) {
        int row = mt * 16 + e + ((lane >> 4) << 3);
        float val = o[mt][nt].f[e] / rowL[row];
        Zg[(size_t)(batch * 1024 + qbase + row) * ld + head * 256 + d] = f2bf(val);
      }
    }
}

// ---------------------------------------------------------------------------
// 6) LayerNorm stats, deterministic two-stage reduction
// ---------------------------------------------------------------------------
__global__ __launch_bounds__(256)
void k_ln_partial(const float* __restrict__ pre, float* __restrict__ partial) {
  __shared__ float sh[256], sh2[256];
  const int b = blockIdx.x >> 6, slice = blockIdx.x & 63;
  const size_t per = 8388608, chunk = per / 64;
  const float* p = pre + (size_t)b * per + (size_t)slice * chunk;
  float s = 0.0f, s2 = 0.0f;
  for (size_t i = threadIdx.x; i < chunk; i += 256) {
    float v = p[i]; s += v; s2 += v * v;
  }
  sh[threadIdx.x] = s; sh2[threadIdx.x] = s2;
  __syncthreads();
  for (int st = 128; st > 0; st >>= 1) {
    if ((int)threadIdx.x < st) {
      sh[threadIdx.x] += sh[threadIdx.x + st];
      sh2[threadIdx.x] += sh2[threadIdx.x + st];
    }
    __syncthreads();
  }
  if (threadIdx.x == 0) {
    partial[blockIdx.x * 2 + 0] = sh[0];
    partial[blockIdx.x * 2 + 1] = sh2[0];
  }
}

__global__ __launch_bounds__(64)
void k_ln_final(const float* __restrict__ partial, float* __restrict__ stats) {
  __shared__ float sh[64], sh2[64];
  const int b = blockIdx.x;
  sh[threadIdx.x]  = partial[(b * 64 + threadIdx.x) * 2 + 0];
  sh2[threadIdx.x] = partial[(b * 64 + threadIdx.x) * 2 + 1];
  __syncthreads();
  for (int st = 32; st > 0; st >>= 1) {
    if ((int)threadIdx.x < st) {
      sh[threadIdx.x] += sh[threadIdx.x + st];
      sh2[threadIdx.x] += sh2[threadIdx.x + st];
    }
    __syncthreads();
  }
  if (threadIdx.x == 0) {
    float inv_n = 1.0f / 8388608.0f;
    float mean = sh[0] * inv_n;
    float var  = sh2[0] * inv_n - mean * mean;
    stats[b * 2 + 0] = mean;
    stats[b * 2 + 1] = rsqrtf(var + 1e-6f);
  }
}

// ---------------------------------------------------------------------------
// 7) normalize * gamma + beta, permute 'b (h w) (n c) -> b n c h w'
// ---------------------------------------------------------------------------
__global__ __launch_bounds__(256)
void k_finalize(const float* __restrict__ pre, const float* __restrict__ stats,
                const float* __restrict__ gamma, const float* __restrict__ beta,
                float* __restrict__ out) {
  int idx = blockIdx.x * blockDim.x + threadIdx.x;
  int w  = idx & 31;
  int h  = (idx >> 5) & 31;
  int cc = (idx >> 10) & 1023;
  int n  = (idx >> 20) & 7;
  int b  = idx >> 23;
  int t = h * 32 + w;
  int f = n * 1024 + cc;
  float mean = stats[b * 2 + 0], inv = stats[b * 2 + 1];
  size_t zi = ((size_t)b * 1024 + t) * 8192 + f;
  size_t gi = (size_t)t * 8192 + f;
  out[idx] = (pre[zi] - mean) * inv * gamma[gi] + beta[gi];
}

// ---------------------------------------------------------------------------
extern "C" void kernel_launch(void* const* d_in, const int* in_sizes, int n_in,
                              void* d_out, int out_size, void* d_ws, size_t ws_size,
                              hipStream_t stream) {
  const float* ref   = (const float*)d_in[0];
  const float* hd    = (const float*)d_in[1];
  const float* wq    = (const float*)d_in[2];
  const float* wk    = (const float*)d_in[3];
  const float* wv    = (const float*)d_in[4];
  const float* wo    = (const float*)d_in[5];
  const float* gamma = (const float*)d_in[6];
  const float* beta  = (const float*)d_in[7];
  float* out = (float*)d_out;

  char* ws = (char*)d_ws;
  size_t off = 0;
  auto alloc = [&](size_t bytes) -> char* {
    char* p = ws + off;
    off += (bytes + 255) & ~(size_t)255;
    return p;
  };
  const size_t XN = 4ull * 1024 * 8192;
  const size_t WN = 8192ull * 2048;
  const size_t QN = 4096ull * 2048;
  float* xf    = (float*)alloc(XN * 4);
  u16*   xbf   = (u16*)  alloc(XN * 2);
  u16*   wqT   = (u16*)  alloc(WN * 2);   // [2048][8192] bf16
  u16*   wkT   = (u16*)  alloc(WN * 2);
  u16*   wvT   = (u16*)  alloc(WN * 2);
  u16*   woT   = (u16*)  alloc(WN * 2);   // [8192][2048] bf16
  u16*   Qb    = (u16*)  alloc(QN * 2);
  u16*   Kb    = (u16*)  alloc(QN * 2);
  u16*   Vb    = (u16*)  alloc(QN * 2);
  u16*   Zb    = (u16*)  alloc(QN * 2);
  float* pre   = (float*)alloc(XN * 4);
  float* part  = (float*)alloc(256 * 2 * 4);
  float* stats = (float*)alloc(8 * 4);

  // 1) weights -> bf16, transposed to [N][K] so GEMM B-tiles are TDM 2D tiles
  dim3 gt1(8192 / 32, 2048 / 32);
  k_convT<<<gt1, 256, 0, stream>>>(wq, wqT, 8192, 2048);
  k_convT<<<gt1, 256, 0, stream>>>(wk, wkT, 8192, 2048);
  k_convT<<<gt1, 256, 0, stream>>>(wv, wvT, 8192, 2048);
  dim3 gt2(2048 / 32, 8192 / 32);
  k_convT<<<gt2, 256, 0, stream>>>(wo, woT, 2048, 8192);

  // 2) x = concat+permute+posenc
  k_prep<<<XN / 256, 256, 0, stream>>>(ref, hd, xf, xbf);

  // 3) Q/K/V projections: [4096,8192] @ [8192,2048] -> bf16
  dim3 gq(2048 / 128, 4096 / 128);
  k_wmma_gemm<true><<<gq, 256, 0, stream>>>(xbf, wqT, Qb, nullptr, nullptr, 4096, 2048, 8192);
  k_wmma_gemm<true><<<gq, 256, 0, stream>>>(xbf, wkT, Kb, nullptr, nullptr, 4096, 2048, 8192);
  k_wmma_gemm<true><<<gq, 256, 0, stream>>>(xbf, wvT, Vb, nullptr, nullptr, 4096, 2048, 8192);

  // 4) fused flash attention -> Zb bf16 [4096, 2048]
  dim3 ga(32, 8, 4);
  k_flash_attn<<<ga, 128, 0, stream>>>(Qb, Kb, Vb, Zb);

  // 5) output projection + residual -> pre fp32
  dim3 go(8192 / 128, 4096 / 128);
  k_wmma_gemm<false><<<go, 256, 0, stream>>>(Zb, woT, nullptr, pre, xf, 4096, 8192, 2048);

  // 6) LayerNorm stats
  k_ln_partial<<<256, 256, 0, stream>>>(pre, part);
  k_ln_final<<<4, 64, 0, stream>>>(part, stats);

  // 7) normalize + gamma/beta + output permute
  k_finalize<<<XN / 256, 256, 0, stream>>>(pre, stats, gamma, beta, out);
}
